// OBBLabelAssigner_39737037423231
// MI455X (gfx1250) — matrix operand order repaired and checked by reading, not compiled, not run on hardware
//
#include <hip/hip_runtime.h>
#include <hip/hip_bf16.h>

// OBB label assigner for MI455X (gfx1250, wave32).
// Heavy pair math (rotated IoU) on VALU; the two genuine GEMMs
// (point-frame rotation, softmax-probability gather) run on the WMMA pipe:
//   - V_WMMA_F32_16X16X4_F32  : ox/oy = [px,py,1] x [c,s,-(cx c+cy s)] (exact f32)
//   - V_WMMA_F32_16X16X32_F16 : softmax(probs) x onehot(gt_labels)      (exact gather)

#define NCLS 15
#define TOPK_K 15
#define BG_LBL 15
#define PI_F 3.14159265358979323846f
#define PI2_F 1.57079632679489661923f

typedef __attribute__((ext_vector_type(2)))  float    v2f;
typedef __attribute__((ext_vector_type(8)))  float    v8f;
typedef __attribute__((ext_vector_type(16))) _Float16 v16h;

// ---------------------------------------------------------------- K0: gt prep
// tab layout (24 floats/gt): [0]cx [1]cy [2]w [3]h [4]ang [5]cos [6]sin [7]area
//                            [8..15] corners x0,y0..x3,y3
__global__ void k_prep_gt(const float* __restrict__ gtb, const int* __restrict__ gtl,
                          float* __restrict__ tab, int* __restrict__ lab,
                          float* __restrict__ Bgeo, int* __restrict__ counts, int G) {
  int g = threadIdx.x;
  if (g >= G) return;
  float cx = gtb[g * 5 + 0], cy = gtb[g * 5 + 1];
  float w  = gtb[g * 5 + 2], h  = gtb[g * 5 + 3], ang = gtb[g * 5 + 4];
  float c = cosf(ang), s = sinf(ang);
  float* t = tab + g * 24;
  t[0] = cx; t[1] = cy; t[2] = w; t[3] = h; t[4] = ang;
  t[5] = c;  t[6] = s;  t[7] = w * h;
  float w2 = 0.5f * w, h2 = 0.5f * h;
  const float dx[4] = { w2, w2, -w2, -w2 };
  const float dy[4] = { -h2, h2, h2, -h2 };
  for (int i = 0; i < 4; ++i) {
    t[8 + 2 * i] = cx + dx[i] * c - dy[i] * s;
    t[9 + 2 * i] = cy + dx[i] * s + dy[i] * c;
  }
  lab[g] = gtl[g];
  counts[g] = 0;
  // Bgeo[4][64]: cols 0..31 -> ox coefficients, cols 32..63 -> oy coefficients
  // ox = px*c + py*s - (cx*c + cy*s)   oy = -px*s + py*c + (cx*s - cy*c)
  Bgeo[0 * 64 + g]      =  c;  Bgeo[1 * 64 + g]      =  s;
  Bgeo[2 * 64 + g]      = -(cx * c + cy * s);
  Bgeo[3 * 64 + g]      =  0.f;
  Bgeo[0 * 64 + 32 + g] = -s;  Bgeo[1 * 64 + 32 + g] =  c;
  Bgeo[2 * 64 + 32 + g] =  (cx * s - cy * c);
  Bgeo[3 * 64 + 32 + g] =  0.f;
}

// ---------------------------------------------------------------- K1: softmax
__global__ void k_softmax(const float* __restrict__ probs, float* __restrict__ psm, int N) {
  int n = blockIdx.x * blockDim.x + threadIdx.x;
  if (n >= N) return;
  const float* p = probs + (size_t)n * NCLS;
  float m = p[0];
  for (int i = 1; i < NCLS; ++i) m = fmaxf(m, p[i]);
  float e[NCLS], sum = 0.f;
  for (int i = 0; i < NCLS; ++i) { e[i] = expf(p[i] - m); sum += e[i]; }
  float inv = 1.f / sum;
  float* o = psm + (size_t)n * 16;
  for (int i = 0; i < NCLS; ++i) o[i] = e[i] * inv;
  o[15] = 0.f;  // K-padding column
}

// --------------------------------------------------- K2: ox/oy via WMMA (f32)
// A (16x4 f32): lanes 0-15 hold row M=lane {K0,K1}; lanes 16-31 hold {K2,K3}.
// B (4x16 f32): lanes 0-15 hold col N=lane {K0,K1}; lanes 16-31 hold {K2,K3}.
// D (16x16 f32): VGPR r, lanes 0-15 -> M=r; lanes 16-31 -> M=r+8; N=lane&15.
__global__ void k_geo_wmma(const float* __restrict__ pts, const float* __restrict__ Bgeo,
                           float* __restrict__ OX, float* __restrict__ OY, int N) {
  int wave = (blockIdx.x * blockDim.x + threadIdx.x) >> 5;
  int lane = threadIdx.x & 31;
  int tiles = (N + 15) >> 4;
  if (wave >= tiles) return;          // wave-uniform: EXEC all-ones inside
  int n0 = wave << 4;
  int half = lane >> 4;               // 0: lanes 0-15, 1: lanes 16-31
  int l = lane & 15;
  bool full = (n0 + 16 <= N);         // wave-uniform full-tile fast path

  v2f a;
  if (half == 0) {
    int nr = n0 + l; if (nr >= N) nr = N - 1;
    a.x = pts[(size_t)nr * 2 + 0];    // K0 = px
    a.y = pts[(size_t)nr * 2 + 1];    // K1 = py
  } else {
    a.x = 1.0f;                       // K2 = 1
    a.y = 0.0f;                       // K3 = 0 (pad)
  }

  for (int j = 0; j < 4; ++j) {
    int col = j * 16 + l;             // 0..63
    v2f b;
    b.x = Bgeo[(half * 2 + 0) * 64 + col];
    b.y = Bgeo[(half * 2 + 1) * 64 + col];
    v8f c = {};
    v8f d = __builtin_amdgcn_wmma_f32_16x16x4_f32(false, a, false, b, (short)0, c,
                                                  false, false);
    float* dst = (col < 32) ? (OX + col) : (OY + (col - 32));
    size_t row0 = (size_t)(n0 + half * 8) * 32;
    if (full) {
#pragma unroll
      for (int r = 0; r < 8; ++r) dst[row0 + (size_t)r * 32] = d[r];
    } else {
      for (int r = 0; r < 8; ++r) {
        int n = n0 + r + half * 8;
        if (n < N) dst[(size_t)n * 32] = d[r];
      }
    }
  }
}

// --------------------------------- K3: softmax-prob gather via WMMA f16 K=32
// A (16x32 f16): lanes 0-15 row M=lane, elems -> K {0..7,16..23};
//                lanes 16-31 row M=lane-16, elems -> K {8..15,24..31}.
// B (32x16 f16): lanes 0-15 col N=lane-16*0, elems -> K {0..7,16..23}; lanes 16-31 K {8..15,24..31}.
__global__ void k_gather_wmma(const float* __restrict__ psm, const int* __restrict__ lab,
                              float* __restrict__ GP, int N, int G) {
  int wave = (blockIdx.x * blockDim.x + threadIdx.x) >> 5;
  int lane = threadIdx.x & 31;
  int tiles = (N + 15) >> 4;
  if (wave >= tiles) return;
  int n0 = wave << 4;
  int half = lane >> 4;
  int l = lane & 15;
  bool full = (n0 + 16 <= N);

  int nr = n0 + l; if (nr >= N) nr = N - 1;
  const float* prow = psm + (size_t)nr * 16;
  v16h a;
  for (int e = 0; e < 16; ++e) {
    int K = (half == 0) ? ((e < 8) ? e : e + 8) : ((e < 8) ? e + 8 : e + 16);
    float v = (K < 16) ? prow[K] : 0.f;   // classes live in K 0..14
    a[e] = (_Float16)v;
  }

  for (int j = 0; j < 2; ++j) {
    int col = j * 16 + l;               // gt index 0..31
    int lbl = (col < G) ? lab[col] : -1;
    v16h b;
    for (int e = 0; e < 16; ++e) {
      int K = ((e < 8) ? e : e + 8) + half * 8;
      b[e] = (K == lbl) ? (_Float16)1.0f : (_Float16)0.0f;
    }
    v8f c = {};
    v8f d = __builtin_amdgcn_wmma_f32_16x16x32_f16(false, a, false, b, (short)0, c,
                                                   false, false);
    float* dst = GP + col;
    size_t row0 = (size_t)(n0 + half * 8) * 32;
    if (full) {
#pragma unroll
      for (int r = 0; r < 8; ++r) dst[row0 + (size_t)r * 32] = d[r];
    } else {
      for (int r = 0; r < 8; ++r) {
        int n = n0 + r + half * 8;
        if (n < N) dst[(size_t)n * 32] = d[r];
      }
    }
  }
}

// ---------------------------------------------------- rotated IoU (per pair)
__device__ __forceinline__ float riou_pair(const float c1x[4], const float c1y[4],
                                           float area1, float cxd, float cyd,
                                           float wd, float hd, float cd, float sd,
                                           const float* __restrict__ t) {
  float c2x[4], c2y[4];
  for (int j = 0; j < 4; ++j) { c2x[j] = t[8 + 2 * j]; c2y[j] = t[9 + 2 * j]; }
  float gcx = t[0], gcy = t[1], gw = t[2], gh = t[3], gc = t[5], gs = t[6];
  float area2 = t[7];

  float px[24], py[24];
  int ok[24];
  // 16 edge-edge intersection candidates
  for (int i = 0; i < 4; ++i) {
    int i1 = (i + 1) & 3;
    float d1x = c1x[i1] - c1x[i], d1y = c1y[i1] - c1y[i];
    for (int j = 0; j < 4; ++j) {
      int j1 = (j + 1) & 3;
      float d2x = c2x[j1] - c2x[j], d2y = c2y[j1] - c2y[j];
      float den = d1x * d2y - d1y * d2x;
      bool okd = fabsf(den) > 1e-10f;
      float safe = okd ? den : 1.f;
      float qpx = c2x[j] - c1x[i], qpy = c2y[j] - c1y[i];
      float tt = (qpx * d2y - qpy * d2x) / safe;
      float uu = (qpx * d1y - qpy * d1x) / safe;
      bool o = okd && tt >= 0.f && tt <= 1.f && uu >= 0.f && uu <= 1.f;
      int idx = i * 4 + j;
      px[idx] = c1x[i] + tt * d1x;
      py[idx] = c1y[i] + tt * d1y;
      ok[idx] = o ? 1 : 0;
    }
  }
  // det vertices inside gt box
  for (int i = 0; i < 4; ++i) {
    float rx = c1x[i] - gcx, ry = c1y[i] - gcy;
    float xx = rx * gc + ry * gs, yy = -rx * gs + ry * gc;
    ok[16 + i] = (fabsf(xx) <= 0.5f * gw + 1e-6f) && (fabsf(yy) <= 0.5f * gh + 1e-6f);
    px[16 + i] = c1x[i]; py[16 + i] = c1y[i];
  }
  // gt vertices inside det box
  for (int j = 0; j < 4; ++j) {
    float rx = c2x[j] - cxd, ry = c2y[j] - cyd;
    float xx = rx * cd + ry * sd, yy = -rx * sd + ry * cd;
    ok[20 + j] = (fabsf(xx) <= 0.5f * wd + 1e-6f) && (fabsf(yy) <= 0.5f * hd + 1e-6f);
    px[20 + j] = c2x[j]; py[20 + j] = c2y[j];
  }

  int cnt = 0; float sx = 0.f, sy = 0.f;
  for (int i = 0; i < 24; ++i)
    if (ok[i]) { ++cnt; sx += px[i]; sy += py[i]; }
  int m = cnt > 1 ? cnt : 1;
  float ctx = sx / (float)m, cty = sy / (float)m;

  float ang[24];
  for (int i = 0; i < 24; ++i) {
    if (ok[i]) ang[i] = atan2f(py[i] - cty, px[i] - ctx);
    else { px[i] = ctx; py[i] = cty; ang[i] = 1e9f; }
  }
  // stable insertion sort ascending by angle (matches stable argsort)
  for (int i = 1; i < 24; ++i) {
    float a = ang[i], x = px[i], y = py[i];
    int j = i - 1;
    while (j >= 0 && ang[j] > a) {
      ang[j + 1] = ang[j]; px[j + 1] = px[j]; py[j + 1] = py[j]; --j;
    }
    ang[j + 1] = a; px[j + 1] = x; py[j + 1] = y;
  }
  float inter = 0.f;
  for (int i = 0; i < m; ++i) {
    int j = (i + 1) % m;
    float vax = px[i] - ctx, vay = py[i] - cty;
    float vbx = px[j] - ctx, vby = py[j] - cty;
    inter += vax * vby - vay * vbx;
  }
  inter = 0.5f * fabsf(inter);
  return inter / fmaxf(area1 + area2 - inter, 1e-8f);
}

// -------------------------------------------- K4: per-pair cost / centerness
__global__ void k_pair(const float* __restrict__ pts, const float* __restrict__ rr,
                       const float* __restrict__ spp, const float* __restrict__ preds,
                       const float* __restrict__ tab, const float* __restrict__ OX,
                       const float* __restrict__ OY, const float* __restrict__ GP,
                       float* __restrict__ COST, float* __restrict__ CENT,
                       unsigned char* __restrict__ TKM, int N, int G) {
  int n = blockIdx.x * blockDim.x + threadIdx.x;
  if (n >= N) return;
  __builtin_prefetch(OX + (size_t)n * 32, 0, 1);
  __builtin_prefetch(OY + (size_t)n * 32, 0, 1);
  float px = pts[(size_t)n * 2], py = pts[(size_t)n * 2 + 1];
  float st = spp[n];
  float r0 = rr[(size_t)n * 2], r1 = rr[(size_t)n * 2 + 1];
  // decode det rbbox (preds scaled by stride, angle unscaled)
  float d0 = preds[(size_t)n * 5 + 0] * st, d1 = preds[(size_t)n * 5 + 1] * st;
  float d2 = preds[(size_t)n * 5 + 2] * st, d3 = preds[(size_t)n * 5 + 3] * st;
  float an = preds[(size_t)n * 5 + 4];
  float ca = cosf(an), sa = sinf(an);
  float wx = d0 + d2, wy = d1 + d3;
  float ot0 = 0.5f * (d2 - d0), ot1 = 0.5f * (d3 - d1);
  float cxd = px + ca * ot0 - sa * ot1;
  float cyd = py + sa * ot0 + ca * ot1;
  float aa = an + PI2_F;
  aa = aa - floorf(aa / PI_F) * PI_F;   // python-style mod -> [0, pi)
  aa -= PI2_F;
  float cd = cosf(aa), sd = sinf(aa);
  float w2d = 0.5f * wx, h2d = 0.5f * wy;
  float c1x[4], c1y[4];
  {
    const float dx[4] = { w2d, w2d, -w2d, -w2d };
    const float dy[4] = { -h2d, h2d, h2d, -h2d };
    for (int i = 0; i < 4; ++i) {
      c1x[i] = cxd + dx[i] * cd - dy[i] * sd;
      c1y[i] = cyd + dx[i] * sd + dy[i] * cd;
    }
  }
  float area1 = wx * wy;
  float ctr_r = 1.5f * st;

  for (int g = 0; g < G; ++g) {
    const float* t = tab + g * 24;
    float w = t[2], h = t[3];
    float ox = OX[(size_t)n * 32 + g], oy = OY[(size_t)n * 32 + g];
    float bt0 = 0.5f * w + ox, bt1 = 0.5f * h + oy;
    float bt2 = 0.5f * w - ox, bt3 = 0.5f * h - oy;
    float mn = fminf(fminf(bt0, bt1), fminf(bt2, bt3));
    float mx = fmaxf(fmaxf(bt0, bt1), fmaxf(bt2, bt3));
    bool inside_gt = mn > 0.f;
    float nx = 2.f * ox / w, ny = 2.f * oy / h;
    float cent = fmaxf(1.f - sqrtf((nx * nx + ny * ny + 1e-8f) * 0.5f), 0.f);
    bool inside_rr = (mx >= r0) && (mx <= r1);
    inside_gt = inside_gt && (fabsf(ox) < ctr_r) && (fabsf(oy) < ctr_r);
    bool valid = inside_gt && inside_rr;
    float iou = riou_pair(c1x, c1y, area1, cxd, cyd, wx, wy, cd, sd, t);
    float p = GP[(size_t)n * 32 + g];
    float cost = (0.2f * cent + 0.2f * iou + 0.6f * p) * (valid ? 1.f : 0.f);
    COST[(size_t)n * 32 + g] = cost;
    CENT[(size_t)n * 32 + g] = cent;
    TKM[(size_t)n * 32 + g] = 0;
  }
}

// ----------------------------------------------- K5: per-gt top-k (15 rounds)
__global__ void k_topk(const float* __restrict__ COST, unsigned char* __restrict__ TKM,
                       int N, int G) {
  int g = blockIdx.x;
  int t = threadIdx.x;
  __shared__ float sv[256];
  __shared__ int   si[256];
  __shared__ int   chosen[TOPK_K];
  __shared__ int   sdone;
  if (t == 0) sdone = 0;
  __syncthreads();
  for (int r = 0; r < TOPK_K; ++r) {
    float bv = -1.f; int bi = N;
    for (int n = t; n < N; n += 256) {
      bool skip = false;
      for (int q = 0; q < r; ++q)
        if (chosen[q] == n) { skip = true; break; }
      if (skip) continue;
      float v = COST[(size_t)n * 32 + g];
      if (v > bv) { bv = v; bi = n; }    // ascending scan -> first max kept
    }
    sv[t] = bv; si[t] = bi;
    __syncthreads();
    for (int s = 128; s > 0; s >>= 1) {
      if (t < s) {
        if (sv[t + s] > sv[t] || (sv[t + s] == sv[t] && si[t + s] < si[t])) {
          sv[t] = sv[t + s]; si[t] = si[t + s];
        }
      }
      __syncthreads();
    }
    if (t == 0) {
      chosen[r] = si[0];
      if (sv[0] > 0.f) TKM[(size_t)si[0] * 32 + g] = 1;  // vals != 0 condition
      else sdone = 1;
    }
    __syncthreads();
    if (sdone) break;
  }
}

// -------------------------------------------------------- K6: area argmax
__global__ void k_assign(const float* __restrict__ tab, const unsigned char* __restrict__ TKM,
                         const int* __restrict__ lab, int* __restrict__ INDS,
                         int* __restrict__ LABS, int* __restrict__ counts, int N, int G) {
  int n = blockIdx.x * blockDim.x + threadIdx.x;
  if (n >= N) return;
  // topk mask implies valid (cost>0 only when valid), so areas = area_g * mask
  float best = 0.f; int bg = 0;
  for (int g = 0; g < G; ++g) {
    float a = TKM[(size_t)n * 32 + g] ? tab[g * 24 + 7] : 0.f;
    if (a > best) { best = a; bg = g; }   // first max, like jnp.argmax
  }
  INDS[n] = bg;
  LABS[n] = (best == 0.f) ? BG_LBL : lab[bg];
  if (best != 0.f) atomicAdd(&counts[bg], 1);
}

// -------------------------------------------- K7: per-gt centerness argmax
__global__ void k_anchor(const float* __restrict__ CENT, int* __restrict__ anchor,
                         int N, int G) {
  int g = blockIdx.x;
  int t = threadIdx.x;
  __shared__ float sv[256];
  __shared__ int   si[256];
  float bv = -1e30f; int bi = 0;
  for (int n = t; n < N; n += 256) {
    float v = CENT[(size_t)n * 32 + g];
    if (v > bv) { bv = v; bi = n; }
  }
  sv[t] = bv; si[t] = bi;
  __syncthreads();
  for (int s = 128; s > 0; s >>= 1) {
    if (t < s) {
      if (sv[t + s] > sv[t] || (sv[t + s] == sv[t] && si[t + s] < si[t])) {
        sv[t] = sv[t + s]; si[t] = si[t + s];
      }
    }
    __syncthreads();
  }
  if (t == 0) anchor[g] = si[0];
}

// ---------------------------------- K8: rescue unmatched gts (sequential g)
__global__ void k_fixup(const int* __restrict__ counts, const int* __restrict__ anchor,
                        const int* __restrict__ lab, int* __restrict__ INDS,
                        int* __restrict__ LABS, int G) {
  if (threadIdx.x != 0 || blockIdx.x != 0) return;
  for (int g = 0; g < G; ++g) {      // ascending -> last-writer-wins on collisions
    if (counts[g] == 0) {
      int a = anchor[g];
      INDS[a] = g;
      LABS[a] = lab[g];
    }
  }
}

// -------------------------------------------------------------- K9: outputs
// d_out layout (floats): [0,N) labels, [N,5N) bbox targets (N x 4), [5N,6N) angle
__global__ void k_final(const float* __restrict__ tab, const float* __restrict__ OX,
                        const float* __restrict__ OY, const float* __restrict__ spp,
                        const int* __restrict__ INDS, const int* __restrict__ LABS,
                        float* __restrict__ out, int N) {
  int n = blockIdx.x * blockDim.x + threadIdx.x;
  if (n >= N) return;
  int g = INDS[n];
  const float* t = tab + g * 24;
  float w = t[2], h = t[3], ang = t[4];
  float ox = OX[(size_t)n * 32 + g], oy = OY[(size_t)n * 32 + g];
  float inv = 1.f / spp[n];
  out[n] = (float)LABS[n];
  out[N + (size_t)n * 4 + 0] = (0.5f * w + ox) * inv;
  out[N + (size_t)n * 4 + 1] = (0.5f * h + oy) * inv;
  out[N + (size_t)n * 4 + 2] = (0.5f * w - ox) * inv;
  out[N + (size_t)n * 4 + 3] = (0.5f * h - oy) * inv;
  out[(size_t)5 * N + n] = ang;
}

extern "C" void kernel_launch(void* const* d_in, const int* in_sizes, int n_in,
                              void* d_out, int out_size, void* d_ws, size_t ws_size,
                              hipStream_t stream) {
  (void)n_in; (void)out_size; (void)ws_size;
  const float* pts = (const float*)d_in[0];
  const float* rr  = (const float*)d_in[1];
  const float* spp = (const float*)d_in[2];
  const float* gtb = (const float*)d_in[3];
  const int*   gtl = (const int*)d_in[4];
  const float* bp  = (const float*)d_in[5];
  const float* pr  = (const float*)d_in[6];
  int N = in_sizes[0] / 2;
  int G = in_sizes[3] / 5;

  char* w = (char*)d_ws;
  auto alloc = [&](size_t bytes) -> void* {
    void* p = (void*)w;
    w += (bytes + 255) & ~((size_t)255);
    return p;
  };
  float* tab    = (float*)alloc((size_t)G * 24 * sizeof(float));
  int*   lab    = (int*)alloc((size_t)G * sizeof(int));
  float* Bgeo   = (float*)alloc(4 * 64 * sizeof(float));
  int*   counts = (int*)alloc((size_t)G * sizeof(int));
  int*   anchor = (int*)alloc((size_t)G * sizeof(int));
  float* psm    = (float*)alloc((size_t)N * 16 * sizeof(float));
  float* OX     = (float*)alloc((size_t)N * 32 * sizeof(float));
  float* OY     = (float*)alloc((size_t)N * 32 * sizeof(float));
  float* GP     = (float*)alloc((size_t)N * 32 * sizeof(float));
  float* COST   = (float*)alloc((size_t)N * 32 * sizeof(float));
  float* CENT   = (float*)alloc((size_t)N * 32 * sizeof(float));
  unsigned char* TKM = (unsigned char*)alloc((size_t)N * 32);
  int*   INDS   = (int*)alloc((size_t)N * sizeof(int));
  int*   LABS   = (int*)alloc((size_t)N * sizeof(int));

  int nb    = (N + 255) / 256;
  int tiles = (N + 15) / 16;
  int wb    = (tiles + 7) / 8;   // 8 wave32 per 256-thread block

  k_prep_gt<<<1, 32, 0, stream>>>(gtb, gtl, tab, lab, Bgeo, counts, G);
  k_softmax<<<nb, 256, 0, stream>>>(pr, psm, N);
  k_geo_wmma<<<wb, 256, 0, stream>>>(pts, Bgeo, OX, OY, N);
  k_gather_wmma<<<wb, 256, 0, stream>>>(psm, lab, GP, N, G);
  k_pair<<<nb, 256, 0, stream>>>(pts, rr, spp, bp, tab, OX, OY, GP, COST, CENT, TKM, N, G);
  k_topk<<<G, 256, 0, stream>>>(COST, TKM, N, G);
  k_assign<<<nb, 256, 0, stream>>>(tab, TKM, lab, INDS, LABS, counts, N, G);
  k_anchor<<<G, 256, 0, stream>>>(CENT, anchor, N, G);
  k_fixup<<<1, 1, 0, stream>>>(counts, anchor, lab, INDS, LABS, G);
  k_final<<<nb, 256, 0, stream>>>(tab, OX, OY, spp, INDS, LABS, (float*)d_out, N);
}
